// DoubleConv_9131100471830
// MI455X (gfx1250) — matrix operand order repaired and checked by reading, not compiled, not run on hardware
//
#include <hip/hip_runtime.h>
#include <hip/hip_bf16.h>
#include <math.h>
#include <stdint.h>

// ---------------------------------------------------------------------------
// CDNA5 (gfx1250, wave32) implementation.
// All matmuls + all 3x3 convs (as implicit GEMM) go through one WMMA kernel:
//   C[M,N] = A[M,K] * W[N,K]^T   (f32 in, f16 WMMA, f32 accum)
// Weight panel is streamed global->LDS with GLOBAL_LOAD_ASYNC_TO_LDS_B128
// (double-buffered, ASYNCcnt-synchronized) whenever the panel is interior.
// ---------------------------------------------------------------------------

typedef __attribute__((ext_vector_type(16))) _Float16 v16h;
typedef __attribute__((ext_vector_type(8)))  float    v8f;

union FragH { v16h v; _Float16 h[16]; };

__device__ __forceinline__ int imin(int a, int b) { return a < b ? a : b; }
__device__ __forceinline__ int imax(int a, int b) { return a > b ? a : b; }

// Issue one per-lane async 16B copy global -> LDS (tracked by ASYNCcnt).
__device__ __forceinline__ void async_ld_b128(uint32_t lds_addr, const float* g)
{
    asm volatile("global_load_async_to_lds_b128 %0, %1, off"
                 :: "v"(lds_addr), "v"(g) : "memory");
}

// ============================ WMMA GEMM =====================================
// a_mode: 0 = A row-major [M,K] with row stride lda
//         1 = im2col of NCHW input (Cin,Hh,Ww), K = Cin*9, M = B*Hh*Ww
// store_mode: 0 = C row-major [M,N]; 1 = NCHW scatter out[b, n, y, x]
// act: 0 none, 1 relu, 2 softplus
__global__ void __launch_bounds__(256)
gemm_wmma_f16(const float* __restrict__ A, const float* __restrict__ Wt,
              float* __restrict__ C, const float* __restrict__ bias,
              const float* __restrict__ resid,
              int M, int N, int K, int lda,
              int a_mode, int store_mode, int Cin, int Hh, int Ww, int act)
{
    __shared__ _Float16 sA[128 * 32];      // 8 KB, f16 staged A tile
    __shared__ float    sW[2][64 * 32];    // 16 KB, double-buffered f32 W tile

    const int tid  = threadIdx.x;
    const int lane = tid & 31;
    const int wave = tid >> 5;          // 0..7
    const int wm   = (wave & 3) * 32;   // 4 waves along M
    const int wn   = (wave >> 2) * 32;  // 2 waves along N
    const int mBase = blockIdx.x * 128;
    const int nBase = blockIdx.y * 64;

    const int  ktiles    = (K + 31) >> 5;
    // async path legal when every W-tile element is in-bounds (uniform per block)
    const bool use_async = (nBase + 64 <= N) && ((K & 31) == 0);

    v8f acc[2][2];
#pragma unroll
    for (int i = 0; i < 2; ++i)
#pragma unroll
        for (int j = 0; j < 2; ++j)
#pragma unroll
            for (int e = 0; e < 8; ++e) acc[i][j][e] = 0.f;

    // ---- weight-panel stagers ----
    auto stage_w_async = [&](int k0, int buf) {
        const uint32_t base = (uint32_t)(uintptr_t)(&sW[buf][0]);
#pragma unroll
        for (int j = 0; j < 2; ++j) {
            const int cid = tid + j * 256;       // 0..511 16B chunks
            const int row = cid >> 3;            // 0..63
            const int col = (cid & 7) * 4;       // 0..28
            const float* g = Wt + (size_t)(nBase + row) * K + (k0 + col);
            async_ld_b128(base + (uint32_t)cid * 16u, g);
        }
    };
    auto stage_w_sync = [&](int k0, int buf) {
        for (int i = tid; i < 64 * 32; i += 256) {
            const int row = i >> 5, col = i & 31;
            const int gn = nBase + row, gk = k0 + col;
            sW[buf][i] = (gn < N && gk < K) ? Wt[(size_t)gn * K + gk] : 0.f;
        }
    };

    // prologue: stage W tile 0 into buffer 0
    if (use_async) stage_w_async(0, 0);
    else           stage_w_sync(0, 0);

    for (int kt = 0; kt < ktiles; ++kt) {
        const int k0 = kt << 5;
        // ---- stage A tile (128x32) as f16 through VGPRs ----
        for (int i = tid; i < 128 * 32; i += 256) {
            const int row = i >> 5, col = i & 31;
            const int gm = mBase + row, gk = k0 + col;
            float v = 0.f;
            if (gk < K && gm < M) {
                if (a_mode == 0) {
                    v = A[(size_t)gm * lda + gk];
                    if (gk + 32 < K) __builtin_prefetch(&A[(size_t)gm * lda + gk + 32], 0, 1);
                } else {
                    const int xw = gm % Ww;
                    const int t1 = gm / Ww;
                    const int yh = t1 % Hh;
                    const int bb = t1 / Hh;
                    const int ci = gk / 9;
                    const int rs = gk - ci * 9;
                    const int r  = rs / 3 - 1;
                    const int s  = rs - (rs / 3) * 3 - 1;
                    const int iy = yh + r, ix = xw + s;
                    if (iy >= 0 && iy < Hh && ix >= 0 && ix < Ww)
                        v = A[(((size_t)bb * Cin + ci) * Hh + iy) * Ww + ix];
                }
            }
            sA[i] = (_Float16)v;
        }
        // ---- prefetch W tile kt+1 into the other buffer ----
        const bool have_next = (kt + 1 < ktiles);
        if (have_next) {
            if (use_async) stage_w_async(k0 + 32, (kt + 1) & 1);
            else           stage_w_sync(k0 + 32, (kt + 1) & 1);
        }
        // current W tile's async copies must have landed (<=2 outstanding
        // belong to the next tile); then make LDS visible workgroup-wide
        if (use_async) {
            if (have_next) asm volatile("s_wait_asynccnt 0x2" ::: "memory");
            else           asm volatile("s_wait_asynccnt 0x0" ::: "memory");
        }
        __syncthreads();

        // ---- build fragments per ISA 16x16x32 f16 layouts ----
        const float* wcur = &sW[kt & 1][0];
        FragH af[2], bf[2];
        const int mlo = lane & 15;
        const int hi  = lane >> 4;
#pragma unroll
        for (int sm = 0; sm < 2; ++sm) {
            const int m  = wm + sm * 16 + mlo;
            const int c0 = hi * 8;
#pragma unroll
            for (int q = 0; q < 8; ++q) af[sm].h[q]     = sA[m * 32 + c0 + q];
#pragma unroll
            for (int q = 0; q < 8; ++q) af[sm].h[8 + q] = sA[m * 32 + c0 + 16 + q];
        }
#pragma unroll
        for (int sn = 0; sn < 2; ++sn) {
            const int n  = wn + sn * 16 + mlo;
            const int kb = hi * 8;
            const float* wp = wcur + n * 32 + kb;
#pragma unroll
            for (int q = 0; q < 8; ++q) bf[sn].h[q] = (_Float16)wp[q];
#pragma unroll
            for (int q = 0; q < 8; ++q) bf[sn].h[8 + q] = (_Float16)wp[16 + q];
        }
#pragma unroll
        for (int sm = 0; sm < 2; ++sm)
#pragma unroll
            for (int sn = 0; sn < 2; ++sn)
                acc[sm][sn] = __builtin_amdgcn_wmma_f32_16x16x32_f16(
                    false, af[sm].v, false, bf[sn].v, (short)0, acc[sm][sn],
                    false, false);
        __syncthreads();   // protect sA and the buffer being refilled next iter
    }

    // ---- epilogue ----
    const int mlo = lane & 15;
    const int hi  = lane >> 4;
#pragma unroll
    for (int sm = 0; sm < 2; ++sm) {
#pragma unroll
        for (int sn = 0; sn < 2; ++sn) {
            const int n = nBase + wn + sn * 16 + mlo;
            if (n >= N) continue;
            const float bv = bias ? bias[n] : 0.f;
#pragma unroll
            for (int r = 0; r < 8; ++r) {
                const int m = mBase + wm + sm * 16 + hi * 8 + r;
                if (m >= M) continue;
                float v = acc[sm][sn][r] + bv;
                if (act == 1) v = fmaxf(v, 0.f);
                else if (act == 2) v = (v > 20.f) ? v : log1pf(expf(v));
                size_t oidx;
                if (store_mode == 0) {
                    oidx = (size_t)m * N + n;
                } else {
                    const int xw = m % Ww;
                    const int t1 = m / Ww;
                    const int yh = t1 % Hh;
                    const int bb = t1 / Hh;
                    oidx = (((size_t)bb * N + n) * Hh + yh) * Ww + xw;
                }
                if (resid) v += resid[oidx];
                C[oidx] = v;
            }
        }
    }
}

// ======================== bilinear resize (align_corners) ===================
__global__ void resize_ac_kernel(const float* __restrict__ in, float* __restrict__ out,
                                 const float* __restrict__ resid,
                                 int BC, int IH, int IW, int OH, int OW)
{
    const int idx = blockIdx.x * 256 + threadIdx.x;
    const int total = BC * OH * OW;
    if (idx >= total) return;
    const int ox = idx % OW;
    const int t  = idx / OW;
    const int oy = t % OH;
    const int bc = t / OH;
    const float sy = (OH > 1) ? (float)(IH - 1) / (float)(OH - 1) : 0.f;
    const float sx = (OW > 1) ? (float)(IW - 1) / (float)(OW - 1) : 0.f;
    const float py = oy * sy, px = ox * sx;
    int y0 = imin((int)floorf(py), IH - 2); y0 = imax(y0, 0);
    int x0 = imin((int)floorf(px), IW - 2); x0 = imax(x0, 0);
    const float fy = py - (float)y0, fx = px - (float)x0;
    const float* p = in + (size_t)bc * IH * IW;
    const float a = p[y0 * IW + x0]       * (1.f - fy) + p[(y0 + 1) * IW + x0]     * fy;
    const float b = p[y0 * IW + x0 + 1]   * (1.f - fy) + p[(y0 + 1) * IW + x0 + 1] * fy;
    float v = a * (1.f - fx) + b * fx;
    if (resid) v += resid[idx];
    out[idx] = v;
}

// ======================== layernorm over last-dim 64 ========================
__global__ void layernorm64_kernel(const float* __restrict__ in, float* __restrict__ out,
                                   const float* __restrict__ g, const float* __restrict__ b,
                                   int rows)
{
    const int wid  = (blockIdx.x * blockDim.x + threadIdx.x) >> 5;
    const int lane = threadIdx.x & 31;
    if (wid >= rows) return;
    const float* p = in + (size_t)wid * 64;
    const float v0 = p[lane], v1 = p[lane + 32];
    float s = v0 + v1, s2 = v0 * v0 + v1 * v1;
#pragma unroll
    for (int o = 16; o > 0; o >>= 1) {
        s  += __shfl_xor(s,  o, 32);
        s2 += __shfl_xor(s2, o, 32);
    }
    const float m   = s * (1.f / 64.f);
    const float var = s2 * (1.f / 64.f) - m * m;
    const float inv = rsqrtf(var + 1e-5f);
    out[(size_t)wid * 64 + lane]      = (v0 - m) * inv * g[lane]      + b[lane];
    out[(size_t)wid * 64 + lane + 32] = (v1 - m) * inv * g[lane + 32] + b[lane + 32];
}

// ======================== softmax over last-dim 64 (inplace) ================
__global__ void softmax64_kernel(float* __restrict__ x, int rows)
{
    const int wid  = (blockIdx.x * blockDim.x + threadIdx.x) >> 5;
    const int lane = threadIdx.x & 31;
    if (wid >= rows) return;
    float* p = x + (size_t)wid * 64;
    float v0 = p[lane], v1 = p[lane + 32];
    float mx = fmaxf(v0, v1);
#pragma unroll
    for (int o = 16; o > 0; o >>= 1) mx = fmaxf(mx, __shfl_xor(mx, o, 32));
    v0 = expf(v0 - mx); v1 = expf(v1 - mx);
    float s = v0 + v1;
#pragma unroll
    for (int o = 16; o > 0; o >>= 1) s += __shfl_xor(s, o, 32);
    const float inv = 1.f / s;
    p[lane] = v0 * inv; p[lane + 32] = v1 * inv;
}

// ============== depthwise causal conv1d (dc=4) + SiLU =======================
__global__ void dwconv_silu_kernel(const float* __restrict__ xz,
                                   const float* __restrict__ w,
                                   const float* __restrict__ bias,
                                   float* __restrict__ xc, int B, int L, int di)
{
    const int idx = blockIdx.x * 256 + threadIdx.x;
    if (idx >= B * L * di) return;
    const int c = idx % di;
    const int t = idx / di;
    const int l = t % L;
    const int b = t / L;
    float acc = bias[c];
#pragma unroll
    for (int k = 0; k < 4; ++k) {
        const int ll = l + k - 3;
        if (ll >= 0)
            acc += xz[((size_t)(b * L + ll) * 2 * di) + c] * w[c * 4 + k];
    }
    xc[idx] = acc * (1.f / (1.f + expf(-acc)));  // silu
}

// ============== selective scan (Mamba recurrence, ds=16) ====================
__global__ void sel_scan_kernel(const float* __restrict__ dt,
                                const float* __restrict__ xdbl, int xrs, int boff, int coff,
                                const float* __restrict__ xc,
                                const float* __restrict__ xz,
                                const float* __restrict__ A_log,
                                const float* __restrict__ Dp,
                                float* __restrict__ yg, int B, int L, int di)
{
    const int idx = blockIdx.x * 256 + threadIdx.x;
    if (idx >= B * di) return;
    const int i = idx % di;
    const int b = idx / di;
    float a[16], h[16];
#pragma unroll
    for (int s = 0; s < 16; ++s) { a[s] = -expf(A_log[i * 16 + s]); h[s] = 0.f; }
    const float Dv = Dp[i];
    for (int l = 0; l < L; ++l) {
        const size_t row = (size_t)(b * L + l);
        const float dtv = dt[row * di + i];
        const float xv  = xc[row * di + i];
        const float zv  = xz[row * 2 * di + di + i];
        const float* xr = xdbl + row * xrs;
        float y = 0.f;
#pragma unroll
        for (int s = 0; s < 16; ++s) {
            h[s] = h[s] * expf(dtv * a[s]) + dtv * xr[boff + s] * xv;
            y += h[s] * xr[coff + s];
        }
        const float yv  = y + Dv * xv;
        const float sig = 1.f / (1.f + expf(-zv));
        yg[row * di + i] = yv * (zv * sig);
    }
}

// ============== LDS-tiled transpose [B,R,C] -> [B,C,R] ======================
__global__ void transpose_kernel(const float* __restrict__ in, float* __restrict__ out,
                                 int R, int C)
{
    __shared__ float tile[32][33];
    const int c0 = blockIdx.x * 32, r0 = blockIdx.y * 32, b = blockIdx.z;
    for (int j = threadIdx.y; j < 32; j += 8) {
        const int r = r0 + j, c = c0 + threadIdx.x;
        tile[j][threadIdx.x] = (r < R && c < C) ? in[((size_t)b * R + r) * C + c] : 0.f;
    }
    __syncthreads();
    for (int j = threadIdx.y; j < 32; j += 8) {
        const int c = c0 + j, r = r0 + threadIdx.x;
        if (c < C && r < R) out[((size_t)b * C + c) * R + r] = tile[threadIdx.x][j];
    }
}

// ============== elementwise multiply ========================================
__global__ void mul_kernel(const float* __restrict__ a, const float* __restrict__ b,
                           float* __restrict__ o, int n)
{
    const int i = blockIdx.x * 256 + threadIdx.x;
    if (i < n) o[i] = a[i] * b[i];
}

// ============== batchnorm (training stats) ==================================
__global__ void zero_kernel(float* p, int n)
{
    const int i = blockIdx.x * 256 + threadIdx.x;
    if (i < n) p[i] = 0.f;
}

__global__ void bn_reduce_kernel(const float* __restrict__ x, float* __restrict__ sum,
                                 float* __restrict__ sumsq, int C, int Bn, int HW)
{
    __shared__ float rs[8], rs2[8];
    const int c = blockIdx.x;
    const size_t per = (size_t)Bn * HW;
    float s = 0.f, s2 = 0.f;
    for (size_t t = (size_t)blockIdx.y * 256 + threadIdx.x; t < per;
         t += (size_t)gridDim.y * 256) {
        const int b = (int)(t / HW);
        const int i = (int)(t % HW);
        const float v = x[((size_t)b * C + c) * HW + i];
        s += v; s2 += v * v;
    }
#pragma unroll
    for (int o = 16; o > 0; o >>= 1) {
        s  += __shfl_xor(s,  o, 32);
        s2 += __shfl_xor(s2, o, 32);
    }
    const int wv = threadIdx.x >> 5, ln = threadIdx.x & 31;
    if (ln == 0) { rs[wv] = s; rs2[wv] = s2; }
    __syncthreads();
    if (threadIdx.x == 0) {
        float a = 0.f, b2 = 0.f;
        for (int i = 0; i < 8; ++i) { a += rs[i]; b2 += rs2[i]; }
        atomicAdd(&sum[c], a);
        atomicAdd(&sumsq[c], b2);
    }
}

__global__ void bn_apply_relu_kernel(float* __restrict__ x, const float* __restrict__ sum,
                                     const float* __restrict__ sumsq,
                                     const float* __restrict__ g, const float* __restrict__ bta,
                                     int C, int Bn, int HW)
{
    const int idx = blockIdx.x * 256 + threadIdx.x;
    if (idx >= Bn * C * HW) return;
    const int c = (idx / HW) % C;
    const float n   = (float)Bn * (float)HW;
    const float m   = sum[c] / n;
    const float var = sumsq[c] / n - m * m;
    x[idx] = fmaxf((x[idx] - m) * rsqrtf(var + 1e-5f) * g[c] + bta[c], 0.f);
}

// ===========================================================================
// Host-side orchestration
// ===========================================================================
static inline void launch_gemm(hipStream_t s, const float* A, const float* Wt, float* C,
                               const float* bias, const float* resid,
                               int M, int N, int K, int lda,
                               int a_mode, int store_mode, int Cin, int H, int W, int act)
{
    dim3 g((M + 127) / 128, (N + 63) / 64);
    gemm_wmma_f16<<<g, 256, 0, s>>>(A, Wt, C, bias, resid, M, N, K, lda,
                                    a_mode, store_mode, Cin, H, W, act);
}

static void run_mamba(hipStream_t s, const float* in, float* out,
                      const float* in_proj, const float* conv_w, const float* conv_b,
                      const float* x_proj, const float* dt_w, const float* dt_b,
                      const float* A_log, const float* Dp, const float* out_proj,
                      int B, int L, int Dm, int di, int dtr,
                      float* xz, float* xc, float* xdbl, float* dtb, float* yg)
{
    const int M = B * L;
    // in_proj: [M,Dm] x [2di,Dm]^T -> xz [M,2di]
    launch_gemm(s, in, in_proj, xz, nullptr, nullptr, M, 2 * di, Dm, Dm, 0, 0, 0, 0, 0, 0);
    // depthwise causal conv + silu -> xc [M,di]
    {
        const int total = M * di;
        dwconv_silu_kernel<<<(total + 255) / 256, 256, 0, s>>>(xz, conv_w, conv_b, xc, B, L, di);
    }
    // x_proj: [M,di] x [dtr+32,di]^T -> xdbl
    launch_gemm(s, xc, x_proj, xdbl, nullptr, nullptr, M, dtr + 32, di, di, 0, 0, 0, 0, 0, 0);
    // dt = softplus(dt_raw @ dt_w^T + b): A = xdbl[:, :dtr]
    launch_gemm(s, xdbl, dt_w, dtb, dt_b, nullptr, M, di, dtr, dtr + 32, 0, 0, 0, 0, 0, 2);
    // selective scan -> yg [M,di]
    {
        const int total = B * di;
        sel_scan_kernel<<<(total + 255) / 256, 256, 0, s>>>(
            dtb, xdbl, dtr + 32, dtr, dtr + 16, xc, xz, A_log, Dp, yg, B, L, di);
    }
    // out_proj: [M,di] x [Dm,di]^T -> out [M,Dm]
    launch_gemm(s, yg, out_proj, out, nullptr, nullptr, M, Dm, di, di, 0, 0, 0, 0, 0, 0);
}

extern "C" void kernel_launch(void* const* d_in, const int* in_sizes, int n_in,
                              void* d_out, int out_size, void* d_ws, size_t ws_size,
                              hipStream_t stream)
{
    // ----- input mapping (setup_inputs dict order, params flattened) -----
    const float* x          = (const float*)d_in[0];   // [4,64,128,128]
    const float* convb1_w   = (const float*)d_in[1];
    const float* convb1_b   = (const float*)d_in[2];
    const float* convb2_w   = (const float*)d_in[3];
    const float* convb2_b   = (const float*)d_in[4];
    const float* ln_g       = (const float*)d_in[5];
    const float* ln_b       = (const float*)d_in[6];
    // m1 (d_model=64, di=128, dtr=4)
    const float* m1_in_proj = (const float*)d_in[7];
    const float* m1_conv_w  = (const float*)d_in[8];
    const float* m1_conv_b  = (const float*)d_in[9];
    const float* m1_x_proj  = (const float*)d_in[10];
    const float* m1_dt_w    = (const float*)d_in[11];
    const float* m1_dt_b    = (const float*)d_in[12];
    const float* m1_A_log   = (const float*)d_in[13];
    const float* m1_D       = (const float*)d_in[14];
    const float* m1_out_proj= (const float*)d_in[15];
    // m2 same shapes as m1
    const float* m2_in_proj = (const float*)d_in[16];
    const float* m2_conv_w  = (const float*)d_in[17];
    const float* m2_conv_b  = (const float*)d_in[18];
    const float* m2_x_proj  = (const float*)d_in[19];
    const float* m2_dt_w    = (const float*)d_in[20];
    const float* m2_dt_b    = (const float*)d_in[21];
    const float* m2_A_log   = (const float*)d_in[22];
    const float* m2_D       = (const float*)d_in[23];
    const float* m2_out_proj= (const float*)d_in[24];
    // m3 (d_model=4096, di=8192, dtr=256)
    const float* m3_in_proj = (const float*)d_in[25];
    const float* m3_conv_w  = (const float*)d_in[26];
    const float* m3_conv_b  = (const float*)d_in[27];
    const float* m3_x_proj  = (const float*)d_in[28];
    const float* m3_dt_w    = (const float*)d_in[29];
    const float* m3_dt_b    = (const float*)d_in[30];
    const float* m3_A_log   = (const float*)d_in[31];
    const float* m3_D       = (const float*)d_in[32];
    const float* m3_out_proj= (const float*)d_in[33];
    const float* smooth_w   = (const float*)d_in[34];
    const float* smooth_b   = (const float*)d_in[35];
    const float* dc1_w      = (const float*)d_in[36];
    const float* bn1_g      = (const float*)d_in[37];
    const float* bn1_b      = (const float*)d_in[38];
    const float* dc2_w      = (const float*)d_in[39];
    const float* bn2_g      = (const float*)d_in[40];
    const float* bn2_b      = (const float*)d_in[41];

    float* out = (float*)d_out;  // [4,128,128,128]
    (void)in_sizes; (void)n_in; (void)out_size; (void)ws_size;

    // ----- workspace carve-up (floats); total ~32.3M floats (~130 MB) -----
    float* W = (float*)d_ws;
    size_t off = 0;
    auto alloc = [&](size_t n) { float* p = W + off; off += n; return p; };
    float* inp  = alloc(4 * 64 * 64 * 64);     // resized input   [4,64,64,64]
    float* t1   = alloc(4 * 16 * 64 * 64);     // convb1 out      [4,16,64,64]
    float* tb   = alloc(4 * 64 * 64 * 64);     // UVMB 't'
    float* xl   = alloc(16384 * 64);           // layernorm out   [4,4096,64]
    float* mout = alloc(16384 * 64);           // m1 out, reused as smooth out
    float* y1   = alloc(4 * 64 * 4096);        // transposed m1, reused as att*z
    float* m3o  = alloc(4 * 64 * 4096);        // m3 out
    float* zb   = alloc(16384 * 64);           // z = transpose(m3 out)
    float* att  = alloc(16384 * 64);           // m2 out -> softmax
    float* up   = alloc(4 * 64 * 128 * 128);   // upsampled + residual
    float* c1   = alloc((size_t)4 * 128 * 128 * 128);  // dc1 out
    float* stats = alloc(256);                 // bn sum[128], sumsq[128]
    // mamba scratch arena (sized for the max of m1/m2/m3)
    float* axz  = alloc(4194304);   // [M, 2di]
    float* axc  = alloc(2097152);   // [M, di]
    float* axdb = alloc(589824);    // [M, dtr+32]
    float* adt  = alloc(2097152);   // [M, di]
    float* ayg  = alloc(2097152);   // [M, di]

    // ---- 1. resize 128->64 (align_corners) ----
    resize_ac_kernel<<<(4 * 64 * 64 * 64 + 255) / 256, 256, 0, stream>>>(
        x, inp, nullptr, 4 * 64, 128, 128, 64, 64);

    // ---- 2. convb1 (64->16, bias, relu) implicit-GEMM WMMA ----
    launch_gemm(stream, inp, convb1_w, t1, convb1_b, nullptr,
                4 * 64 * 64, 16, 64 * 9, 0, 1, 1, 64, 64, 64, 1);
    // ---- 3. convb2 (16->64, bias) + residual(inp) ----
    launch_gemm(stream, t1, convb2_w, tb, convb2_b, inp,
                4 * 64 * 64, 64, 16 * 9, 0, 1, 1, 16, 64, 64, 0);

    // ---- 4. layernorm over raw-reshaped [16384, 64] ----
    layernorm64_kernel<<<(16384 * 32 + 255) / 256, 256, 0, stream>>>(tb, xl, ln_g, ln_b, 16384);

    // ---- 5. mamba m1 on xl [4, 4096, 64] ----
    run_mamba(stream, xl, mout, m1_in_proj, m1_conv_w, m1_conv_b, m1_x_proj,
              m1_dt_w, m1_dt_b, m1_A_log, m1_D, m1_out_proj,
              4, 4096, 64, 128, 4, axz, axc, axdb, adt, ayg);

    // ---- 6. y1 = swapaxes(mout,1,2): [4,4096,64] -> [4,64,4096] ----
    {
        dim3 g((64 + 31) / 32, (4096 + 31) / 32, 4);
        transpose_kernel<<<g, dim3(32, 8), 0, stream>>>(mout, y1, 4096, 64);
    }

    // ---- 7. mamba m3 on y1 [4, 64, 4096] (the big one) ----
    run_mamba(stream, y1, m3o, m3_in_proj, m3_conv_w, m3_conv_b, m3_x_proj,
              m3_dt_w, m3_dt_b, m3_A_log, m3_D, m3_out_proj,
              4, 64, 4096, 8192, 256, axz, axc, axdb, adt, ayg);

    // ---- 8. z = swapaxes(m3o,1,2): [4,64,4096] -> [4,4096,64] ----
    {
        dim3 g((4096 + 31) / 32, (64 + 31) / 32, 4);
        transpose_kernel<<<g, dim3(32, 8), 0, stream>>>(m3o, zb, 64, 4096);
    }

    // ---- 9. mamba m2 on xl -> att, then softmax over last dim ----
    run_mamba(stream, xl, att, m2_in_proj, m2_conv_w, m2_conv_b, m2_x_proj,
              m2_dt_w, m2_dt_b, m2_A_log, m2_D, m2_out_proj,
              4, 4096, 64, 128, 4, axz, axc, axdb, adt, ayg);
    softmax64_kernel<<<(16384 * 32 + 255) / 256, 256, 0, stream>>>(att, 16384);

    // ---- 10. az = att * z (raw-reshape to NCHW), smooth conv 64->64 ----
    mul_kernel<<<(1048576 + 255) / 256, 256, 0, stream>>>(att, zb, y1, 1048576);
    launch_gemm(stream, y1, smooth_w, mout, smooth_b, nullptr,
                4 * 64 * 64, 64, 64 * 9, 0, 1, 1, 64, 64, 64, 0);

    // ---- 11. resize 64->128 + residual x ----
    resize_ac_kernel<<<(4 * 64 * 128 * 128 + 255) / 256, 256, 0, stream>>>(
        mout, up, x, 4 * 64, 64, 64, 128, 128);

    // ---- 12. dc1 conv (64->128, no bias) + BN(train) + ReLU ----
    launch_gemm(stream, up, dc1_w, c1, nullptr, nullptr,
                4 * 128 * 128, 128, 64 * 9, 0, 1, 1, 64, 128, 128, 0);
    zero_kernel<<<1, 256, 0, stream>>>(stats, 256);
    {
        dim3 g(128, 64);
        bn_reduce_kernel<<<g, 256, 0, stream>>>(c1, stats, stats + 128, 128, 4, 128 * 128);
    }
    bn_apply_relu_kernel<<<(8388608 + 255) / 256, 256, 0, stream>>>(
        c1, stats, stats + 128, bn1_g, bn1_b, 128, 4, 128 * 128);

    // ---- 13. dc2 conv (128->128, no bias) -> d_out, + BN + ReLU ----
    launch_gemm(stream, c1, dc2_w, out, nullptr, nullptr,
                4 * 128 * 128, 128, 128 * 9, 0, 1, 1, 128, 128, 128, 0);
    zero_kernel<<<1, 256, 0, stream>>>(stats, 256);
    {
        dim3 g(128, 64);
        bn_reduce_kernel<<<g, 256, 0, stream>>>(out, stats, stats + 128, 128, 4, 128 * 128);
    }
    bn_apply_relu_kernel<<<(8388608 + 255) / 256, 256, 0, stream>>>(
        out, stats, stats + 128, bn2_g, bn2_b, 128, 4, 128 * 128);
}
